// TransducerJoint_35613868818712
// MI455X (gfx1250) — matrix operand order repaired and checked by reading, not compiled, not run on hardware
//
#include <hip/hip_runtime.h>
#include <hip/hip_bf16.h>

// ---------------------------------------------------------------------------
// TransducerJoint for MI455X (gfx1250, wave32, WMMA + async LDS staging)
//   enc  = encoder @ W_enc^T + b_enc          (B*T, H)   fp32  (tiny GEMM)
//   pred = predictor @ W_pred^T + b_pred      (B*U, H)   fp32  (tiny GEMM)
//   out  = LN(relu(enc+pred)) @ W_out^T + b_out  -> (B,T,U,V)  fused kernel
// Fused kernel: bf16 activations staged in LDS, W_out bf16 double-buffered
// into LDS via global_load_async_to_lds_b128 (ASYNCcnt), consumed by
// v_wmma_f32_16x16x32_bf16. 512 threads/WG (16 waves) so the 128 accumulator
// VGPRs per wave fit without scratch spills.
// ---------------------------------------------------------------------------

#define B_    4
#define T_    256
#define U_    128
#define DENC  512
#define DPRED 640
#define H_    512
#define V_    1024

#define A_PITCH 520   // bf16 elements per A row in LDS (1040 B) - bank spread

typedef __attribute__((ext_vector_type(16))) __bf16 v16bf;
typedef __attribute__((ext_vector_type(8)))  __bf16 v8bf;
typedef __attribute__((ext_vector_type(4)))  __bf16 v4bf;
typedef __attribute__((ext_vector_type(8)))  float  v8f;

// ---- async-copy + wait helpers ---------------------------------------------
#if defined(__has_builtin)
#if __has_builtin(__builtin_amdgcn_s_wait_asynccnt)
#define WAIT_ASYNC0() __builtin_amdgcn_s_wait_asynccnt(0)
#endif
#endif
#ifndef WAIT_ASYNC0
#define WAIT_ASYNC0() asm volatile("s_wait_asynccnt 0" ::: "memory")
#endif

__device__ __forceinline__ unsigned lds_addr32(const void* p) {
  // generic LDS pointer: low 32 bits == wave-relative LDS byte address
  return (unsigned)(uintptr_t)p;
}

__device__ __forceinline__ void async_b128(unsigned lds_byte,
                                           unsigned long long gaddr) {
  asm volatile("global_load_async_to_lds_b128 %0, %1, off"
               :: "v"(lds_byte), "v"(gaddr) : "memory");
}

// column `col` of the W tile: copy 64B (32 K-values, bf16) into LDS with a
// 16B-chunk XOR swizzle keyed on (col>>2)&3 so later reads are bank-conflict
// free.
__device__ __forceinline__ void issue_w_col(const __bf16* __restrict__ Wb,
                                            int kk, const __bf16* buf,
                                            int col) {
  const unsigned lbase = lds_addr32(buf) + col * 64;
  const unsigned long long g =
      (unsigned long long)(uintptr_t)((const char*)Wb + (size_t)col * (H_ * 2) + kk * 64);
  const int sw = (col >> 2) & 3;
#pragma unroll
  for (int j = 0; j < 4; ++j)
    async_b128(lbase + ((j ^ sw) << 4), g + j * 16);
}

// ---- W_out fp32 -> bf16 -----------------------------------------------------
__global__ void cvt_bf16_kernel(const float* __restrict__ w,
                                __bf16* __restrict__ wb, int n) {
  int i = blockIdx.x * blockDim.x + threadIdx.x;
  if (i < n) wb[i] = (__bf16)w[i];
}

// ---- small fp32 projection GEMM: Y[M][H] = X[M][D] . W[H][D]^T + bias ------
__global__ void proj_kernel(const float* __restrict__ X,
                            const float* __restrict__ W,
                            const float* __restrict__ bias,
                            float* __restrict__ Y, int D) {
  __shared__ float Xs[16][17];
  __shared__ float Ws[16][17];
  const int tx = threadIdx.x, ty = threadIdx.y;
  const int row = blockIdx.y * 16 + ty;   // m
  const int col = blockIdx.x * 16 + tx;   // h
  float acc = 0.f;
  for (int d0 = 0; d0 < D; d0 += 16) {
    Xs[ty][tx] = X[row * D + d0 + tx];
    Ws[ty][tx] = W[(blockIdx.x * 16 + ty) * D + d0 + tx];
    __syncthreads();
#pragma unroll
    for (int k = 0; k < 16; ++k) acc += Xs[ty][k] * Ws[tx][k];
    __syncthreads();
  }
  Y[row * H_ + col] = acc + bias[col];
}

// ---- fused joint + LayerNorm + output GEMM ---------------------------------
// grid: 2048 WGs x 512 threads (16 waves). WG = (b,t, 64-u block).
// Stage A: 4 LN rows per wave (8 lanes/row) -> bf16 into LDS.
// Stage B: 16 waves = 2 row-groups x 8 col-groups; each wave 32 rows x 128
//          cols = 16 accum tiles; K-loop 16 steps; B from double-buffered
//          LDS filled by async copies.
__global__ __launch_bounds__(512, 1)
void joint_ln_out_kernel(const float* __restrict__ enc,    // (B*T, H)
                         const float* __restrict__ pred,   // (B*U, H)
                         const float* __restrict__ gamma,  // (H)
                         const float* __restrict__ beta,   // (H)
                         const __bf16* __restrict__ Wb,    // (V, H) bf16
                         const float* __restrict__ bout,   // (V)
                         float* __restrict__ out)          // (B*T*U, V)
{
  __shared__ __bf16 As[64 * A_PITCH];   // 66,560 B
  __shared__ __bf16 Bs[2][32 * V_];     // 2 x 65,536 B

  const int tid  = threadIdx.x;
  const int lane = tid & 31;
  const int wave = tid >> 5;            // 0..15
  const int s    = lane & 15;
  const int half = lane >> 4;

  const int g  = blockIdx.x;       // 0..2047
  const int bt = g >> 1;           // 0..B*T-1
  const int u0 = (g & 1) * 64;
  const int b  = bt >> 8;          // T_ = 256

  // kick off async DMA of W tile 0 before doing anything else (2 cols/thread)
  issue_w_col(Wb, 0, &Bs[0][0], tid);
  issue_w_col(Wb, 0, &Bs[0][0], tid + 512);

  // ---------------- Stage A: relu(enc+pred) -> LayerNorm -> bf16 LDS -------
  {
    const int q   = lane >> 3;                          // row within wave
    const int s8  = lane & 7;
    const int row = wave * 4 + q;                       // 0..63
    const float4* encp  = (const float4*)(enc  + (size_t)bt * H_);
    const float4* predp = (const float4*)(pred + (size_t)(b * U_ + u0 + row) * H_);

    float x[64];
    float sum = 0.f, sumsq = 0.f;
#pragma unroll
    for (int j = 0; j < 16; ++j) {
      const int h4 = s8 + j * 8;                        // float4 index, coalesced
      float4 e = encp[h4];
      float4 p = predp[h4];
      float4 v;
      v.x = fmaxf(e.x + p.x, 0.f);
      v.y = fmaxf(e.y + p.y, 0.f);
      v.z = fmaxf(e.z + p.z, 0.f);
      v.w = fmaxf(e.w + p.w, 0.f);
      sum   += v.x + v.y + v.z + v.w;
      sumsq += v.x * v.x + v.y * v.y + v.z * v.z + v.w * v.w;
      x[j * 4 + 0] = v.x; x[j * 4 + 1] = v.y;
      x[j * 4 + 2] = v.z; x[j * 4 + 3] = v.w;
    }
    // 8-lane butterfly (xor 1,2,4 stays within each 8-lane group)
#pragma unroll
    for (int o = 1; o < 8; o <<= 1) {
      sum   += __shfl_xor(sum,   o, 32);
      sumsq += __shfl_xor(sumsq, o, 32);
    }
    const float mean = sum * (1.f / H_);
    const float var  = sumsq * (1.f / H_) - mean * mean;
    const float rs   = rsqrtf(var + 1e-5f);

    const float4* gmp = (const float4*)gamma;
    const float4* btp = (const float4*)beta;
#pragma unroll
    for (int j = 0; j < 16; ++j) {
      const int h4 = s8 + j * 8;
      float4 gm = gmp[h4];
      float4 bb = btp[h4];
      v4bf y;
      y.x = (__bf16)((x[j*4+0] - mean) * rs * gm.x + bb.x);
      y.y = (__bf16)((x[j*4+1] - mean) * rs * gm.y + bb.y);
      y.z = (__bf16)((x[j*4+2] - mean) * rs * gm.z + bb.z);
      y.w = (__bf16)((x[j*4+3] - mean) * rs * gm.w + bb.w);
      *(v4bf*)(As + row * A_PITCH + h4 * 4) = y;        // 8B LDS store
    }
  }

  // ---------------- Stage B: bf16 WMMA GEMM, async double-buffered W -------
  const int rg = wave & 1;          // row group: rows rg*32 .. +31
  const int cg = wave >> 1;         // col group: cols cg*128 .. +127
  const int nb = cg * 128;

  v8f acc[2][8] = {};               // 2 row-tiles x 8 col-tiles of 16x16 f32

  union Frag { struct { v8bf lo, hi; } p; v16bf v; };

  for (int kk = 0; kk < 16; ++kk) {
    WAIT_ASYNC0();                  // my async writes of tile kk landed
    __syncthreads();                // everyone's landed; prev buffer free
    if (kk + 1 < 16) {
      issue_w_col(Wb, kk + 1, &Bs[(kk + 1) & 1][0], tid);        // overlap
      issue_w_col(Wb, kk + 1, &Bs[(kk + 1) & 1][0], tid + 512);  // with WMMAs
    }

    // A fragments: lane half 0 -> K {0..7,16..23}, half 1 -> {8..15,24..31}
    const __bf16* arow0 = As + (rg * 32 + s) * A_PITCH + kk * 32 + half * 8;
    const __bf16* arow1 = arow0 + 16 * A_PITCH;
    Frag a0, a1;
    a0.p.lo = *(const v8bf*)(arow0);
    a0.p.hi = *(const v8bf*)(arow0 + 16);
    a1.p.lo = *(const v8bf*)(arow1);
    a1.p.hi = *(const v8bf*)(arow1 + 16);

    const __bf16* bbase = &Bs[kk & 1][0];
#pragma unroll
    for (int n = 0; n < 8; ++n) {
      const int col = nb + n * 16 + s;
      const int sw  = (col >> 2) & 3;
      const char* cb = (const char*)(bbase + col * 32);
      Frag bf;
      bf.p.lo = *(const v8bf*)(cb + (((half)     ^ sw) << 4));
      bf.p.hi = *(const v8bf*)(cb + (((half + 2) ^ sw) << 4));
      acc[0][n] = __builtin_amdgcn_wmma_f32_16x16x32_bf16(
          false, a0.v, false, bf.v, (short)0, acc[0][n], false, false);
      acc[1][n] = __builtin_amdgcn_wmma_f32_16x16x32_bf16(
          false, a1.v, false, bf.v, (short)0, acc[1][n], false, false);
    }
  }

  // ---------------- Epilogue: + b_out, store ------------------------------
  const size_t row_base = (size_t)bt * U_ + u0;
#pragma unroll
  for (int r = 0; r < 2; ++r) {
#pragma unroll
    for (int n = 0; n < 8; ++n) {
      const int col = nb + n * 16 + s;
      const float bo = bout[col];
#pragma unroll
      for (int i = 0; i < 8; ++i) {
        const int mrow = rg * 32 + r * 16 + half * 8 + i;  // C layout per ISA
        out[(row_base + mrow) * V_ + col] = acc[r][n][i] + bo;
      }
    }
  }
}

// ---------------------------------------------------------------------------
extern "C" void kernel_launch(void* const* d_in, const int* in_sizes, int n_in,
                              void* d_out, int out_size, void* d_ws, size_t ws_size,
                              hipStream_t stream) {
  const float* enc_in  = (const float*)d_in[0];
  const float* pred_in = (const float*)d_in[1];
  const float* W_enc   = (const float*)d_in[2];
  const float* b_enc   = (const float*)d_in[3];
  const float* W_pred  = (const float*)d_in[4];
  const float* b_pred  = (const float*)d_in[5];
  const float* gamma   = (const float*)d_in[6];
  const float* beta    = (const float*)d_in[7];
  const float* W_out   = (const float*)d_in[8];
  const float* b_out   = (const float*)d_in[9];
  float* out = (float*)d_out;

  // workspace layout: enc (2MB f32) | pred (1MB f32) | W_out bf16 (1MB)
  float*  enc_ws  = (float*)d_ws;
  float*  pred_ws = enc_ws + (size_t)B_ * T_ * H_;
  __bf16* Wb      = (__bf16*)(pred_ws + (size_t)B_ * U_ * H_);

  cvt_bf16_kernel<<<(V_ * H_) / 256, 256, 0, stream>>>(W_out, Wb, V_ * H_);

  proj_kernel<<<dim3(H_ / 16, (B_ * T_) / 16), dim3(16, 16), 0, stream>>>(
      enc_in, W_enc, b_enc, enc_ws, DENC);
  proj_kernel<<<dim3(H_ / 16, (B_ * U_) / 16), dim3(16, 16), 0, stream>>>(
      pred_in, W_pred, b_pred, pred_ws, DPRED);

  joint_ln_out_kernel<<<dim3(2 * B_ * T_), dim3(512), 0, stream>>>(
      enc_ws, pred_ws, gamma, beta, Wb, b_out, out);
}
